// SampleModel_82282983456909
// MI455X (gfx1250) — matrix operand order repaired and checked by reading, not compiled, not run on hardware
//
#include <hip/hip_runtime.h>
#include <hip/hip_bf16.h>
#include <math.h>

// ---------------------------------------------------------------------------
// Adaptive softmax NLL (Transformer-XL style) for MI455X / gfx1250.
//
// Dominant work: three fused GEMM+log-softmax passes over the vocab
// (2048x20002x1024 + 2048x40000x256 + 2048x40000x64 ~ 70 GMAC), done with
// v_wmma_f32_16x16x32_bf16 and an online (max, sumexp) reduction so the
// 650MB of logits never touch HBM. Weight streams ~133MB ~= 5.7us @23.3TB/s.
// Two column tiles are processed per wave iteration with independent
// accumulators so bf16-convert VALU work co-executes under the other tile's
// WMMA instead of draining into WMMA->VALU WAR hazard v_nops.
// ---------------------------------------------------------------------------

typedef __attribute__((ext_vector_type(16))) __bf16 v16bf;
typedef __attribute__((ext_vector_type(8)))  float  v8f;

#define LOG2E 1.4426950408889634f
#define NSEQ  2048

union ABfrag { v16bf v; int4 q[2]; };

// ---------------------------------------------------------------------------
// Kernel 1: per-token adaptive embedding (gather + cluster matvec, *sqrt(D))
// followed by the three output projections. One 256-thread block per token.
// h0/h1/h2 are written as bf16 A-operands for the WMMA GEMMs.
// ---------------------------------------------------------------------------
__global__ __launch_bounds__(256)
void embed_project_kernel(
    const int* __restrict__ xs,
    const float* __restrict__ ew0, const float* __restrict__ ew1, const float* __restrict__ ew2,
    const float* __restrict__ ep0, const float* __restrict__ ep1, const float* __restrict__ ep2,
    const float* __restrict__ op0, const float* __restrict__ op1, const float* __restrict__ op2,
    __bf16* __restrict__ h0, __bf16* __restrict__ h1, __bf16* __restrict__ h2)
{
    __shared__ float emb[1024];
    __shared__ float hid[1024];

    const int t   = blockIdx.x;
    const int tid = threadIdx.x;
    const int x   = xs[t];

    const float* erow;
    const float* proj;
    int demb;
    if (x < 20000)      { erow = ew0 + (size_t)x * 1024;          proj = ep0; demb = 1024; }
    else if (x < 60000) { erow = ew1 + (size_t)(x - 20000) * 256; proj = ep1; demb = 256; }
    else                { erow = ew2 + (size_t)(x - 60000) * 64;  proj = ep2; demb = 64; }

    for (int k = tid; k < demb; k += 256) emb[k] = erow[k];
    __syncthreads();

    const float scale = 32.0f;  // sqrt(1024)
    for (int d = tid; d < 1024; d += 256) {
        const float* pr = proj + (size_t)d * demb;   // emb_proj_i[d, :]
        float acc = 0.0f;
        for (int k = 0; k < demb; ++k) acc += emb[k] * pr[k];
        hid[d] = acc * scale;
    }
    __syncthreads();

    // h0 = hidden @ out_proj0   (out_proj0[k][j], coalesced over j)
    for (int j = tid; j < 1024; j += 256) {
        float acc = 0.0f;
        for (int k = 0; k < 1024; ++k) acc += hid[k] * op0[(size_t)k * 1024 + j];
        h0[(size_t)t * 1024 + j] = (__bf16)acc;
    }
    // h1 = hidden @ out_proj1
    {
        int j = tid;  // 256 outputs, 256 threads
        float acc = 0.0f;
        for (int k = 0; k < 1024; ++k) acc += hid[k] * op1[(size_t)k * 256 + j];
        h1[(size_t)t * 256 + j] = (__bf16)acc;
    }
    // h2 = hidden @ out_proj2
    if (tid < 64) {
        int j = tid;
        float acc = 0.0f;
        for (int k = 0; k < 1024; ++k) acc += hid[k] * op2[(size_t)k * 64 + j];
        h2[(size_t)t * 64 + j] = (__bf16)acc;
    }
}

// ---------------------------------------------------------------------------
// Kernel 2: fused GEMM (bf16 WMMA, fp32 accum) + online log-softmax stats.
// Block = 4 waves = 128 threads, owns a 16-row tile of tokens.
// gridDim.y = 4 column segments; each writes partial (max, sumexp) per row.
// Each wave iteration covers TWO 16-column tiles with independent
// accumulators (hazard hiding). Captures the logit at each row's (clipped)
// target column; for the head also columns n1 / n1+1 (cluster tokens).
// ---------------------------------------------------------------------------
template<int KDIM, bool HEAD>
__global__ __launch_bounds__(128)
void fused_gemm_softmax(
    const __bf16* __restrict__ A,     // [2048][KDIM] bf16, row-major
    const float*  __restrict__ W1,    // [n1][KDIM] fp32, row-major (K contig)
    const float*  __restrict__ W2,    // [2][KDIM] cluster_w (HEAD only)
    const float*  __restrict__ B1,    // [n1]
    const float*  __restrict__ B2,    // [2] (HEAD only)
    const int*    __restrict__ labels,
    int n1, int ncols, int lab_off, int lab_hi,
    float* __restrict__ pm,           // [4][2048] partial running max
    float* __restrict__ ps,           // [4][2048] partial running sumexp
    float* __restrict__ capL,         // [2048] logit at clipped target col
    float* __restrict__ cap0,         // [2048] logit at col n1   (HEAD)
    float* __restrict__ cap1)         // [2048] logit at col n1+1 (HEAD)
{
    __shared__ __bf16 Alds[16 * KDIM];
    __shared__ float red_m[4][16];
    __shared__ float red_s[4][16];

    const int tid     = threadIdx.x;
    const int wave    = tid >> 5;
    const int lane    = tid & 31;
    const int rowBase = blockIdx.x * 16;

    // Stage the 16 x KDIM bf16 activation tile into LDS (flat copy).
    {
        const int4* Ag = reinterpret_cast<const int4*>(A + (size_t)rowBase * KDIM);
        int4* Al = reinterpret_cast<int4*>(Alds);
        for (int i = tid; i < 16 * KDIM / 8; i += 128) Al[i] = Ag[i];
    }
    __syncthreads();

    const int colInTile = lane & 15;
    const int halfHi    = (lane >= 16) ? 1 : 0;
    const int koffB     = halfHi ? 16 : 0;   // B: low lanes K0..15, high lanes K16..31
    const int koffA     = halfHi ? 16 : 0;   // A: low lanes K0..15, high lanes K16..31
    const int rowA      = lane & 15;

    // Per-lane target columns for the 8 rows this lane's accumulator covers.
    int capc[8];
    #pragma unroll
    for (int r = 0; r < 8; ++r) {
        int trow = rowBase + r + halfHi * 8;
        int lab  = labels[trow] - lab_off;
        capc[r]  = lab < 0 ? 0 : (lab > lab_hi ? lab_hi : lab);
    }

    float m[8], s[8];
    #pragma unroll
    for (int r = 0; r < 8; ++r) { m[r] = -INFINITY; s[r] = 0.0f; }

    // Weight-row / bias lookup with column clamp (invalid cols masked later).
    auto wsel = [&](int col, const float*& wrow, float& bias) {
        int ccol = (col < ncols) ? col : (ncols - 1);
        if (ccol < n1) { wrow = W1 + (size_t)ccol * KDIM;        bias = B1[ccol]; }
        else           { wrow = W2 + (size_t)(ccol - n1) * KDIM; bias = B2[ccol - n1]; }
    };
    // fp32 weight row chunk -> bf16 B fragment (16 contiguous K values).
    auto loadB = [&](const float* wrow, int kb) {
        const float4* wq = reinterpret_cast<const float4*>(wrow + kb + koffB);
        float4 f0 = wq[0], f1 = wq[1], f2 = wq[2], f3 = wq[3];
        ABfrag b;
        b.v[0]  = (__bf16)f0.x; b.v[1]  = (__bf16)f0.y; b.v[2]  = (__bf16)f0.z; b.v[3]  = (__bf16)f0.w;
        b.v[4]  = (__bf16)f1.x; b.v[5]  = (__bf16)f1.y; b.v[6]  = (__bf16)f1.z; b.v[7]  = (__bf16)f1.w;
        b.v[8]  = (__bf16)f2.x; b.v[9]  = (__bf16)f2.y; b.v[10] = (__bf16)f2.z; b.v[11] = (__bf16)f2.w;
        b.v[12] = (__bf16)f3.x; b.v[13] = (__bf16)f3.y; b.v[14] = (__bf16)f3.z; b.v[15] = (__bf16)f3.w;
        return b;
    };
    // Online softmax update + in-flight capture for one finished tile.
    auto consume = [&](const v8f& acc, int col, float bias) {
        if (col >= ncols) return;
        #pragma unroll
        for (int r = 0; r < 8; ++r) {
            float v = acc[r] + bias;
            if (v > m[r]) { s[r] = s[r] * exp2f((m[r] - v) * LOG2E) + 1.0f; m[r] = v; }
            else          { s[r] += exp2f((v - m[r]) * LOG2E); }
            int trow = rowBase + r + halfHi * 8;
            if (col == capc[r]) capL[trow] = v;
            if constexpr (HEAD) {
                if (col == n1)     cap0[trow] = v;
                if (col == n1 + 1) cap1[trow] = v;
            }
        }
    };

    const int ntiles = (ncols + 15) / 16;
    // 16 wave-slots chip-wide per row-tile (4 waves x 4 column segments),
    // each consuming tile pairs.
    for (int tp = (blockIdx.y * 4 + wave) * 2; tp < ntiles; tp += 32) {
        const int col0 = tp * 16 + colInTile;
        const int col1 = col0 + 16;
        const float* wrow0; float bias0;
        const float* wrow1; float bias1;
        wsel(col0, wrow0, bias0);
        wsel(col1, wrow1, bias1);

        v8f acc0 = {};
        v8f acc1 = {};
        #pragma unroll
        for (int kc = 0; kc < KDIM / 32; ++kc) {
            const int kb = kc * 32;
            ABfrag a;
            a.q[0] = *reinterpret_cast<const int4*>(&Alds[rowA * KDIM + kb + koffA]);
            a.q[1] = *reinterpret_cast<const int4*>(&Alds[rowA * KDIM + kb + koffA + 8]);
            ABfrag b0 = loadB(wrow0, kb);
            ABfrag b1 = loadB(wrow1, kb);
            acc0 = __builtin_amdgcn_wmma_f32_16x16x32_bf16(
                false, a.v, false, b0.v, (short)0, acc0, false, false);
            acc1 = __builtin_amdgcn_wmma_f32_16x16x32_bf16(
                false, a.v, false, b1.v, (short)0, acc1, false, false);
        }
        consume(acc0, col0, bias0);
        consume(acc1, col1, bias1);
    }

    // Butterfly merge across the 16 lanes of each half (rows 0-7 / 8-15).
    #pragma unroll
    for (int r = 0; r < 8; ++r) {
        #pragma unroll
        for (int off = 1; off < 16; off <<= 1) {
            float om = __shfl_xor(m[r], off, 32);
            float os = __shfl_xor(s[r], off, 32);
            if (om > m[r]) { s[r] = s[r] * exp2f((m[r] - om) * LOG2E) + os; m[r] = om; }
            else           { s[r] += os * exp2f((om - m[r]) * LOG2E); }
        }
    }
    if (lane == 0 || lane == 16) {
        #pragma unroll
        for (int r = 0; r < 8; ++r) {
            red_m[wave][r + halfHi * 8] = m[r];
            red_s[wave][r + halfHi * 8] = s[r];
        }
    }
    __syncthreads();

    // Cross-wave merge; emit this segment's partial (max, sumexp) per row.
    if (tid < 16) {
        float M = red_m[0][tid], S = red_s[0][tid];
        #pragma unroll
        for (int w = 1; w < 4; ++w) {
            float om = red_m[w][tid], os = red_s[w][tid];
            if (om > M) { S = S * exp2f((M - om) * LOG2E) + os; M = om; }
            else        { S += os * exp2f((om - M) * LOG2E); }
        }
        pm[blockIdx.y * NSEQ + rowBase + tid] = M;
        ps[blockIdx.y * NSEQ + rowBase + tid] = S;
    }
}

// ---------------------------------------------------------------------------
// Kernel 3: merge the 4 column-segment partials into per-row LSEs and emit
// the NLL according to the label's cluster.
// ---------------------------------------------------------------------------
__global__ void finalize_kernel(
    const int* __restrict__ labels,
    const float* __restrict__ hpm, const float* __restrict__ hps,
    const float* __restrict__ hcapL, const float* __restrict__ hcap0, const float* __restrict__ hcap1,
    const float* __restrict__ t1pm, const float* __restrict__ t1ps, const float* __restrict__ t1cap,
    const float* __restrict__ t2pm, const float* __restrict__ t2ps, const float* __restrict__ t2cap,
    float* __restrict__ out)
{
    int t = blockIdx.x * blockDim.x + threadIdx.x;
    if (t >= NSEQ) return;

    auto merge_lse = [&](const float* pm, const float* ps) -> float {
        float M = pm[t], S = ps[t];
        #pragma unroll
        for (int seg = 1; seg < 4; ++seg) {
            float om = pm[seg * NSEQ + t], os = ps[seg * NSEQ + t];
            if (om > M) { S = S * exp2f((M - om) * LOG2E) + os; M = om; }
            else        { S += os * exp2f((om - M) * LOG2E); }
        }
        return M + logf(S);
    };

    float hlse  = merge_lse(hpm, hps);
    int   lab   = labels[t];
    float lp;
    if (lab < 20000)      lp = hcapL[t] - hlse;                                        // head vocab
    else if (lab < 60000) lp = (hcap1[t] - hlse) + (t1cap[t] - merge_lse(t1pm, t1ps)); // tail 1 (head col -1)
    else                  lp = (hcap0[t] - hlse) + (t2cap[t] - merge_lse(t2pm, t2ps)); // tail 2 (head col -2)
    out[t] = -lp;
}

// ---------------------------------------------------------------------------
extern "C" void kernel_launch(void* const* d_in, const int* in_sizes, int n_in,
                              void* d_out, int out_size, void* d_ws, size_t ws_size,
                              hipStream_t stream) {
    const int*   x        = (const int*)  d_in[0];
    const int*   labels   = (const int*)  d_in[1];
    const float* emb_w0   = (const float*)d_in[2];
    const float* emb_w1   = (const float*)d_in[3];
    const float* emb_w2   = (const float*)d_in[4];
    const float* emb_p0   = (const float*)d_in[5];
    const float* emb_p1   = (const float*)d_in[6];
    const float* emb_p2   = (const float*)d_in[7];
    const float* out_w0   = (const float*)d_in[8];
    const float* out_b0   = (const float*)d_in[9];
    const float* out_w1   = (const float*)d_in[10];
    const float* out_b1   = (const float*)d_in[11];
    const float* out_w2   = (const float*)d_in[12];
    const float* out_b2   = (const float*)d_in[13];
    const float* out_p0   = (const float*)d_in[14];
    const float* out_p1   = (const float*)d_in[15];
    const float* out_p2   = (const float*)d_in[16];
    const float* clus_w   = (const float*)d_in[17];
    const float* clus_b   = (const float*)d_in[18];
    float*       out      = (float*)d_out;

    // Workspace bump allocation (~5.8 MB total).
    char* ws = (char*)d_ws;
    size_t off = 0;
    auto alloc = [&](size_t bytes) { char* p = ws + off; off += (bytes + 255) & ~(size_t)255; return p; };
    __bf16* h0   = (__bf16*)alloc((size_t)NSEQ * 1024 * 2);
    __bf16* h1   = (__bf16*)alloc((size_t)NSEQ * 256 * 2);
    __bf16* h2   = (__bf16*)alloc((size_t)NSEQ * 64 * 2);
    float* hpm   = (float*)alloc(4 * NSEQ * 4);
    float* hps   = (float*)alloc(4 * NSEQ * 4);
    float* t1pm  = (float*)alloc(4 * NSEQ * 4);
    float* t1ps  = (float*)alloc(4 * NSEQ * 4);
    float* t2pm  = (float*)alloc(4 * NSEQ * 4);
    float* t2ps  = (float*)alloc(4 * NSEQ * 4);
    float* hcapL = (float*)alloc(NSEQ * 4);
    float* hcap0 = (float*)alloc(NSEQ * 4);
    float* hcap1 = (float*)alloc(NSEQ * 4);
    float* t1cap = (float*)alloc(NSEQ * 4);
    float* t2cap = (float*)alloc(NSEQ * 4);

    // 1) adaptive embedding + projections
    embed_project_kernel<<<NSEQ, 256, 0, stream>>>(
        x, emb_w0, emb_w1, emb_w2, emb_p0, emb_p1, emb_p2,
        out_p0, out_p1, out_p2, h0, h1, h2);

    // 2) head: [20000 vocab + 2 cluster tokens], K=1024
    fused_gemm_softmax<1024, true><<<dim3(NSEQ / 16, 4), 128, 0, stream>>>(
        h0, out_w0, clus_w, out_b0, clus_b, labels,
        20000, 20002, 0, 19999, hpm, hps, hcapL, hcap0, hcap1);

    //    tail 1: 40000 vocab, K=256
    fused_gemm_softmax<256, false><<<dim3(NSEQ / 16, 4), 128, 0, stream>>>(
        h1, out_w1, out_w1, out_b1, out_b1, labels,
        40000, 40000, 20000, 39999, t1pm, t1ps, t1cap, nullptr, nullptr);

    //    tail 2: 40000 vocab, K=64
    fused_gemm_softmax<64, false><<<dim3(NSEQ / 16, 4), 128, 0, stream>>>(
        h2, out_w2, out_w2, out_b2, out_b2, labels,
        40000, 40000, 60000, 39999, t2pm, t2ps, t2cap, nullptr, nullptr);

    // 3) merge segments, pick cluster, emit NLL
    finalize_kernel<<<(NSEQ + 255) / 256, 256, 0, stream>>>(
        labels, hpm, hps, hcapL, hcap0, hcap1,
        t1pm, t1ps, t1cap, t2pm, t2ps, t2cap, out);
}
